// MultiheadAutoCorrelation_72911364817673
// MI455X (gfx1250) — matrix operand order repaired and checked by reading, not compiled, not run on hardware
//
#include <hip/hip_runtime.h>
#include <hip/hip_bf16.h>

// ---------------------------------------------------------------------------
// CDNA5 (gfx1250) Multihead AutoCorrelation
//   L = S = 2048, B = 8, D = 1024, H = 16, E = 64, TOPK = int(ln 2048) = 7
//
// Pipeline:
//   fp32->bf16 pack  ->  WMMA GEMM projections (Q/K emitted as bf16)
//   -> WMMA Q^T K diagonal-sum correlation (LDS reduction)
//   -> top-7 + softmax -> delayed gather -> WMMA output projection
// GEMM A-tiles are double-buffered into LDS with the Tensor Data Mover
// (tensor_load_to_lds pipelined against WMMA via s_wait_tensorcnt 1).
// ---------------------------------------------------------------------------

typedef __bf16 bf16_t;
typedef __attribute__((ext_vector_type(16))) __bf16 v16bf;
typedef __attribute__((ext_vector_type(8)))  __bf16 v8bf;
typedef __attribute__((ext_vector_type(8)))  float  v8f;
typedef unsigned int u32x4 __attribute__((ext_vector_type(4)));
typedef int          i32x8 __attribute__((ext_vector_type(8)));
typedef int          i32x4 __attribute__((ext_vector_type(4)));

#define L_LEN 2048
#define B_SZ  8
#define D_DIM 1024
#define H_CNT 16
#define E_DIM 64
#define TOPK  7

struct U8 { uint4 a, b; };

__device__ inline v16bf frag_from(uint4 a, uint4 b) {
    U8 u{a, b};
    return __builtin_bit_cast(v16bf, u);
}

// ---------------------------------------------------------------------------
// TDM: issue a 2D tile load (tile_w x tile_h elements, row stride in elements)
// from global to LDS byte offset lds_byte. D# packing per ISA 8.3/8.4:
// group0: count=1, lds_addr, global_addr(57b), type=2; group1: data_size,
// tensor dims == tile dims (always in-bounds), tile dims, dim0 stride.
// ---------------------------------------------------------------------------
__device__ inline void tdm_load_tile_2d(unsigned lds_byte, const void* gtile,
                                        unsigned tile_w, unsigned tile_h,
                                        unsigned long long row_stride_elems,
                                        unsigned data_size_code /*0=1B,1=2B,2=4B*/)
{
    const unsigned long long ga = (unsigned long long)gtile;
    u32x4 g0;
    g0[0] = 1u;                                              // count=1
    g0[1] = lds_byte;                                        // lds_addr
    g0[2] = (unsigned)ga;                                    // global_addr lo
    g0[3] = (unsigned)((ga >> 32) & 0x01ffffffull) | 0x80000000u; // hi + type=2
    i32x8 g1;
    g1[0] = (int)(data_size_code << 16);                     // wg_mask=0, dsize
    g1[1] = (int)((tile_w & 0xffffu) << 16);                 // tensor_dim0 lo16
    g1[2] = (int)(((tile_w >> 16) & 0xffffu) | ((tile_h & 0xffffu) << 16));
    g1[3] = (int)(((tile_h >> 16) & 0xffffu) | ((tile_w & 0xffffu) << 16)); // tile_dim0
    g1[4] = (int)(tile_h & 0xffffu);                         // tile_dim1 (tile_dim2=0)
    g1[5] = (int)(row_stride_elems & 0xffffffffull);         // dim0_stride lo32
    g1[6] = (int)((row_stride_elems >> 32) & 0xffffull);     // dim0_stride hi16
    g1[7] = 0;
    i32x4 z4 = (i32x4){0, 0, 0, 0};
#if defined(__clang_major__) && (__clang_major__ >= 23)
    i32x8 z8 = (i32x8){0, 0, 0, 0, 0, 0, 0, 0};
    __builtin_amdgcn_tensor_load_to_lds(g0, g1, z4, z4, z8, 0);
#else
    __builtin_amdgcn_tensor_load_to_lds(g0, g1, z4, z4, 0);
#endif
}

// ---------------------------------------------------------------------------
// Pack fp32 -> bf16, 8 elements/thread (b128 load pair -> b128 store).
// ---------------------------------------------------------------------------
__global__ __launch_bounds__(256) void cvt_f32_to_bf16(
    const float* __restrict__ x, __bf16* __restrict__ y, unsigned n)
{
    const unsigned i = (blockIdx.x * 256u + threadIdx.x) * 8u;
    if (i >= n) return;
    const float4 a = *(const float4*)(x + i);
    const float4 b = *(const float4*)(x + i + 4);
    v8bf o;
    o[0] = (bf16_t)a.x; o[1] = (bf16_t)a.y; o[2] = (bf16_t)a.z; o[3] = (bf16_t)a.w;
    o[4] = (bf16_t)b.x; o[5] = (bf16_t)b.y; o[6] = (bf16_t)b.z; o[7] = (bf16_t)b.w;
    *(v8bf*)(y + i) = o;
}

// ---------------------------------------------------------------------------
// GEMM: Y[m,n] = sum_k X[m,k]*W[n,k] + bias[n]   (bf16 inputs, f32 accum)
// Block: 256 thr = 8 waves, one 16-row m-tile, 8x64 = 512 columns.
// A-tile (16x32 bf16) double-buffered in LDS by the TDM: tile k+1 is in
// flight (s_wait_tensorcnt 1) while the 4 WMMAs consume tile k.
// ---------------------------------------------------------------------------
template <bool OUT_BF16>
__global__ __launch_bounds__(256) void gemm_bf16_xwt(
    const __bf16* __restrict__ Xb,   // [M,K]
    const __bf16* __restrict__ Wb,   // [N,K]
    const float*  __restrict__ bias, // [N]
    float* __restrict__ Yf, __bf16* __restrict__ Yb,
    int M, int N, int K)
{
    __shared__ unsigned atile[2][16][16];  // 2 x (16 rows x 32 bf16), LDS off 0
    const int lane = threadIdx.x & 31;
    const int wave = threadIdx.x >> 5;
    const int nblk = N >> 9;                      // N / 512
    const int tm   = (blockIdx.x / nblk) << 4;
    const int tn   = ((blockIdx.x % nblk) << 9) + (wave << 6);
    const int lo = lane & 15, hi = lane >> 4;

    v8f acc[4];
    #pragma unroll
    for (int s = 0; s < 4; ++s) {
        const float bv = bias[tn + 16 * s + lo];
        #pragma unroll
        for (int r = 0; r < 8; ++r) acc[s][r] = bv;
    }

    const unsigned Ku = (unsigned)K >> 1;
    const unsigned* __restrict__ br0 =
        (const unsigned*)Wb + (size_t)(tn + lo) * Ku;

    // Prime the pipeline: tile for kb=0 into buffer 0.
    if (wave == 0)
        tdm_load_tile_2d(0u, Xb + (size_t)tm * K, 32u, 16u,
                         (unsigned long long)K, 1u);

    int cur = 0;
    for (int kb = 0; kb < K; kb += 32) {
        if (wave == 0) {                          // wave-uniform; TDM ignores EXEC
            if (kb + 32 < K) {
                tdm_load_tile_2d((unsigned)((cur ^ 1) << 10),
                                 Xb + (size_t)tm * K + kb + 32, 32u, 16u,
                                 (unsigned long long)K, 1u);
                __builtin_amdgcn_s_wait_tensorcnt(1);  // tile kb resident
            } else {
                __builtin_amdgcn_s_wait_tensorcnt(0);
            }
        }
        __syncthreads();                          // publish atile[cur]

        // A 16x32 bf16 frag (ISA 7.12.2): uint idx j<4 -> 4*hi+j ; j>=4 -> 8+4*hi+(j-4)
        const uint4 a0 = *(const uint4*)&atile[cur][lo][4 * hi];
        const uint4 a1 = *(const uint4*)&atile[cur][lo][8 + 4 * hi];
        const v16bf af = frag_from(a0, a1);

        #pragma unroll
        for (int s = 0; s < 4; ++s) {
            // B 32x16 bf16 frag: uint idx j<4 -> 4*hi+j ; j>=4 -> 8+4*hi+(j-4), col = lane&15
            const unsigned* __restrict__ br = br0 + (size_t)s * 16 * Ku + (kb >> 1);
            if (kb + 32 < K) __builtin_prefetch(br + 16, 0, 1);
            const v16bf bf = frag_from(*(const uint4*)(br + 4 * hi), *(const uint4*)(br + 8 + 4 * hi));
            acc[s] = __builtin_amdgcn_wmma_f32_16x16x32_bf16(
                false, af, false, bf, (short)0, acc[s], false, false);
        }
        __syncthreads();                          // reads of atile[cur] done
        cur ^= 1;
    }

    #pragma unroll
    for (int s = 0; s < 4; ++s) {
        const int col = tn + 16 * s + lo;
        #pragma unroll
        for (int r = 0; r < 8; ++r) {
            const int row = tm + r + (hi ? 8 : 0);
            if (OUT_BF16) Yb[(size_t)row * N + col] = (bf16_t)acc[s][r];
            else          Yf[(size_t)row * N + col] = acc[s][r];
        }
    }
}

// ---------------------------------------------------------------------------
// corr[b,h,tau] = sum_t sum_e Q[b,h,e,t] * K[b,h,e,(t-tau) mod L]
// via 16x16 WMMA tiles of the per-head Gram matrix, scatter-reduced by
// (t - s) mod L into an LDS accumulator (never materializes the 2.1 GB
// score tensor). Two independent accumulators (one per e-chunk) avoid the
// WMMA->WMMA RAW hazard; merged with one VALU add at scatter time.
// ---------------------------------------------------------------------------
__global__ __launch_bounds__(256) void autocorr_scores(
    const __bf16* __restrict__ Qb, const __bf16* __restrict__ Kb,
    float* __restrict__ corr)
{
    __shared__ float cl[L_LEN];
    const int bh = blockIdx.x;             // b*16 + h
    const int strip = blockIdx.y;          // 0..15
    const int b = bh >> 4, h = bh & 15;
    const int lane = threadIdx.x & 31;
    const int wave = threadIdx.x >> 5;
    const int lo = lane & 15, hi = lane >> 4;

    for (int i = threadIdx.x; i < L_LEN; i += 256) cl[i] = 0.0f;
    __syncthreads();

    const int t0 = strip * 128 + wave * 16;
    const unsigned* __restrict__ Qu = (const unsigned*)Qb;
    const unsigned* __restrict__ KuP = (const unsigned*)Kb;
    const unsigned colbase = (unsigned)h * 32u;  // (h*64)/2 uints
    const unsigned rowu = (D_DIM / 2);           // 512 uints per (t,b) row

    // A fragments (Q tile), loop-invariant over s
    v16bf af[2];
    #pragma unroll
    for (int c = 0; c < 2; ++c) {
        const unsigned* __restrict__ ar =
            Qu + ((size_t)(t0 + lo) * B_SZ + b) * rowu + colbase + 16 * c;
        af[c] = frag_from(*(const uint4*)(ar + 4 * hi),
                          *(const uint4*)(ar + 8 + 4 * hi));
    }

    for (int s0 = 0; s0 < L_LEN; s0 += 16) {
        const unsigned* __restrict__ krow =
            KuP + ((size_t)(s0 + lo) * B_SZ + b) * rowu + colbase;
        const v16bf bf0 = frag_from(*(const uint4*)(krow + 4 * hi),
                                    *(const uint4*)(krow + 8 + 4 * hi));
        const v16bf bf1 = frag_from(*(const uint4*)(krow + 16 + 4 * hi),
                                    *(const uint4*)(krow + 24 + 4 * hi));
        v8f z;
        #pragma unroll
        for (int r = 0; r < 8; ++r) z[r] = 0.0f;
        // independent WMMAs (no D->C chaining hazard)
        const v8f p0 = __builtin_amdgcn_wmma_f32_16x16x32_bf16(
            false, af[0], false, bf0, (short)0, z, false, false);
        const v8f p1 = __builtin_amdgcn_wmma_f32_16x16x32_bf16(
            false, af[1], false, bf1, (short)0, z, false, false);

        const int m0 = t0 + (hi ? 8 : 0);
        const int s  = s0 + lo;
        #pragma unroll
        for (int r = 0; r < 8; ++r) {
            const int tau = (m0 + r - s) & (L_LEN - 1);
            atomicAdd(&cl[tau], p0[r] + p1[r]);   // ds_add_f32
        }
    }
    __syncthreads();

    float* __restrict__ cg = corr + (size_t)bh * L_LEN;
    for (int i = threadIdx.x; i < L_LEN; i += 256) atomicAdd(&cg[i], cl[i]);
}

// ---------------------------------------------------------------------------
// Per (b,h): top-7 lags of corr/E, softmax weights. One thread per (b,h).
// ---------------------------------------------------------------------------
__global__ void topk_softmax(const float* __restrict__ corr,
                             float* __restrict__ wout, int* __restrict__ tauout)
{
    const int bh = blockIdx.x * blockDim.x + threadIdx.x;
    if (bh >= B_SZ * H_CNT) return;
    const float* __restrict__ c = corr + (size_t)bh * L_LEN;

    float best[TOPK]; int bidx[TOPK];
    #pragma unroll
    for (int i = 0; i < TOPK; ++i) { best[i] = -1e30f; bidx[i] = 0; }

    for (int t = 0; t < L_LEN; ++t) {
        const float v = c[t];
        if (v > best[TOPK - 1]) {
            int p = TOPK - 1;
            while (p > 0 && best[p - 1] < v) {
                best[p] = best[p - 1]; bidx[p] = bidx[p - 1]; --p;
            }
            best[p] = v; bidx[p] = t;
        }
    }
    const float mx = best[0] * (1.0f / (float)E_DIM);
    float e[TOPK], ssum = 0.0f;
    #pragma unroll
    for (int i = 0; i < TOPK; ++i) {
        e[i] = __expf(best[i] * (1.0f / (float)E_DIM) - mx);
        ssum += e[i];
    }
    const float inv = 1.0f / ssum;
    #pragma unroll
    for (int i = 0; i < TOPK; ++i) {
        wout[bh * TOPK + i]   = e[i] * inv;
        tauout[bh * TOPK + i] = bidx[i];
    }
}

// ---------------------------------------------------------------------------
// Y[t,b,d] = (1/TOPK) * sum_k w[b,h,k] * Vp[(t + tau_k) mod L, b, d], h=d/64
// ---------------------------------------------------------------------------
__global__ __launch_bounds__(256) void delayed_agg(
    const float* __restrict__ Vp, const float* __restrict__ wg,
    const int* __restrict__ taug, float* __restrict__ Y)
{
    const size_t idx = (size_t)blockIdx.x * 256 + threadIdx.x;
    const int d  = (int)(idx % D_DIM);
    const size_t tb = idx / D_DIM;
    const int b  = (int)(tb % B_SZ);
    const int t  = (int)(tb / B_SZ);
    const int bh = b * H_CNT + (d >> 6);

    float acc = 0.0f;
    #pragma unroll
    for (int k = 0; k < TOPK; ++k) {
        const int   tau = taug[bh * TOPK + k];
        const float wk  = wg[bh * TOPK + k];
        const int   ts  = (t + tau) & (L_LEN - 1);
        acc += wk * Vp[((size_t)ts * B_SZ + b) * D_DIM + d];
    }
    Y[idx] = acc * (1.0f / (float)TOPK);
}

// ---------------------------------------------------------------------------
extern "C" void kernel_launch(void* const* d_in, const int* in_sizes, int n_in,
                              void* d_out, int out_size, void* d_ws, size_t ws_size,
                              hipStream_t stream)
{
    (void)in_sizes; (void)n_in; (void)out_size; (void)ws_size;
    const float* query = (const float*)d_in[0];
    const float* key   = (const float*)d_in[1];
    const float* value = (const float*)d_in[2];
    const float* Wq    = (const float*)d_in[3];
    const float* bq    = (const float*)d_in[4];
    const float* Wk    = (const float*)d_in[5];
    const float* bk    = (const float*)d_in[6];
    const float* Wv    = (const float*)d_in[7];
    const float* bv    = (const float*)d_in[8];
    const float* Wo    = (const float*)d_in[9];
    const float* bo    = (const float*)d_in[10];
    float* out = (float*)d_out;

    const int M = L_LEN * B_SZ;                 // 16384
    const unsigned mat = (unsigned)M * D_DIM;   // 16.78M elements
    const unsigned dd  = D_DIM * D_DIM;         // 1.05M elements

    float*  Vp = (float*)d_ws;                  // [L,B,D] f32
    float*  Y  = Vp + (size_t)mat;              // [L,B,D] f32
    __bf16* Qb = (__bf16*)(Y + (size_t)mat);    // [L,B,D] bf16
    __bf16* Kb = Qb + (size_t)mat;
    __bf16* Xs = Kb + (size_t)mat;              // bf16 staging: X
    __bf16* Ws = Xs + (size_t)mat;              // bf16 staging: W
    float* corr = (float*)(Ws + (size_t)dd);    // [B*H, L]
    float* wg   = corr + (size_t)B_SZ * H_CNT * L_LEN;
    int*   taug = (int*)(wg + B_SZ * H_CNT * TOPK);

    hipMemsetAsync(corr, 0, (size_t)B_SZ * H_CNT * L_LEN * sizeof(float), stream);

    const int cvtM = mat / 2048, cvtW = dd / 2048;
    const int gemmBlocks = (M / 16) * (D_DIM / 512);   // 2048

    // Q projection -> bf16
    cvt_f32_to_bf16<<<cvtM, 256, 0, stream>>>(query, Xs, mat);
    cvt_f32_to_bf16<<<cvtW, 256, 0, stream>>>(Wq, Ws, dd);
    gemm_bf16_xwt<true><<<gemmBlocks, 256, 0, stream>>>(Xs, Ws, bq, nullptr, Qb,
                                                        M, D_DIM, D_DIM);
    // K projection -> bf16
    cvt_f32_to_bf16<<<cvtM, 256, 0, stream>>>(key, Xs, mat);
    cvt_f32_to_bf16<<<cvtW, 256, 0, stream>>>(Wk, Ws, dd);
    gemm_bf16_xwt<true><<<gemmBlocks, 256, 0, stream>>>(Xs, Ws, bk, nullptr, Kb,
                                                        M, D_DIM, D_DIM);
    // V projection -> f32
    cvt_f32_to_bf16<<<cvtM, 256, 0, stream>>>(value, Xs, mat);
    cvt_f32_to_bf16<<<cvtW, 256, 0, stream>>>(Wv, Ws, dd);
    gemm_bf16_xwt<false><<<gemmBlocks, 256, 0, stream>>>(Xs, Ws, bv, Vp, nullptr,
                                                         M, D_DIM, D_DIM);

    autocorr_scores<<<dim3(B_SZ * H_CNT, 16), 256, 0, stream>>>(Qb, Kb, corr);
    topk_softmax<<<1, 128, 0, stream>>>(corr, wg, taug);
    delayed_agg<<<(int)(mat / 256), 256, 0, stream>>>(Vp, wg, taug, Y);

    // Output projection -> f32 (d_out)
    cvt_f32_to_bf16<<<cvtM, 256, 0, stream>>>(Y, Xs, mat);
    cvt_f32_to_bf16<<<cvtW, 256, 0, stream>>>(Wo, Ws, dd);
    gemm_bf16_xwt<false><<<gemmBlocks, 256, 0, stream>>>(Xs, Ws, bo, out, nullptr,
                                                         M, D_DIM, D_DIM);
}
